// HSAM_63239098466569
// MI455X (gfx1250) — compile-verified, matrix operand order checked by reading, not run on hardware
//
#include <hip/hip_runtime.h>
#include <hip/hip_bf16.h>
#include <math.h>

typedef __bf16 bf16_t;
typedef __attribute__((ext_vector_type(16))) __bf16 v16bf;
typedef __attribute__((ext_vector_type(8)))  float  v8f;

#define NGRAPH 128
#define GK 128          // all GEMMs in this net have K == 128
#define BT_STRIDE 136   // LDS row stride (bf16 elems): 272B = 68 dwords, conflict-free halves

// ---------- helpers: ordered-uint encoding for float atomicMax ----------
__device__ __forceinline__ unsigned int enc_key(float f) {
  unsigned int b = __float_as_uint(f);
  return (b & 0x80000000u) ? ~b : (b | 0x80000000u);
}
__device__ __forceinline__ float dec_key(unsigned int k) {
  unsigned int b = (k & 0x80000000u) ? (k ^ 0x80000000u) : ~k;
  return __uint_as_float(b);
}

// ---------- tiny elementwise kernels ----------
__global__ void k_zero32(unsigned int* p, int n) {
  int i = blockIdx.x * blockDim.x + threadIdx.x;
  if (i < n) p[i] = 0u;
}

__global__ void k_cvt_bf16(const float* __restrict__ a, bf16_t* __restrict__ o, int n) {
  int i = blockIdx.x * blockDim.x + threadIdx.x;
  if (i < n) o[i] = (bf16_t)a[i];
}

__global__ void k_gelu_bf16(const float* __restrict__ a, bf16_t* __restrict__ o, int n) {
  int i = blockIdx.x * blockDim.x + threadIdx.x;
  if (i < n) {
    float x = a[i];
    o[i] = (bf16_t)(0.5f * x * (1.0f + erff(x * 0.70710678118654752f)));
  }
}

// build 128x128 block-diagonal bf16 matrix from per-head [2,64,64] f32 weights
__global__ void k_blockdiag(const float* __restrict__ rel, bf16_t* __restrict__ o) {
  int i = blockIdx.x * blockDim.x + threadIdx.x;   // 16384 threads
  if (i >= 128 * 128) return;
  int r = i >> 7, c = i & 127;
  int hr = r >> 6, hc = c >> 6;
  o[i] = (hr == hc) ? (bf16_t)rel[(hr << 12) + ((r & 63) << 6) + (c & 63)] : (bf16_t)0.0f;
}

// layer-1 projections: input is N x 1, so k/q/v are outer products (no GEMM needed)
__global__ void k_layer1_kqv(const float* __restrict__ x,
                             const float* __restrict__ Wk, const float* __restrict__ bk,
                             const float* __restrict__ Wq, const float* __restrict__ bq,
                             const float* __restrict__ Wv, const float* __restrict__ bv,
                             float* __restrict__ k, float* __restrict__ q, float* __restrict__ v,
                             bf16_t* __restrict__ kbf, bf16_t* __restrict__ vbf, int N) {
  int i = blockIdx.x * blockDim.x + threadIdx.x;
  if (i >= N * 128) return;
  int n = i >> 7, j = i & 127;
  float xv = x[n];
  float kk = xv * Wk[j] + bk[j];
  float qq = xv * Wq[j] + bq[j];
  float vv = xv * Wv[j] + bv[j];
  k[i] = kk; q[i] = qq; v[i] = vv;
  kbf[i] = (bf16_t)kk; vbf[i] = (bf16_t)vv;
}

// ---------- WMMA GEMM: C[M,N] = A[M,128] @ B[128,N] (+bias, +epilogue) ----------
// Per block: fixed 16-col N-tile; B tile (128x16) staged once into LDS, transposed
// to [col][k] so fragment reads are ds_load_b128. All 4 k-step fragments are
// preloaded (two 8-wide load clauses), then the 4 WMMAs issue back-to-back.
// OOB A-rows are clamped (their D rows are discarded by the store guard), so no
// exec-mask branching in the hot path.
// mode 0: bias only; mode 1: bias + relu; mode 2: bias + sigmoid-gated skip + relu
__global__ void k_gemm_wmma(const bf16_t* __restrict__ A, const bf16_t* __restrict__ B,
                            const float* __restrict__ bias,
                            const float* __restrict__ skipx, const float* __restrict__ gate,
                            float* __restrict__ C, bf16_t* __restrict__ Cbf,
                            int M, int N, int mode) {
  __shared__ bf16_t Bt[16 * BT_STRIDE];
  const int tid = threadIdx.x;
  // cooperative load of B tile, transposed into LDS: Bt[col][k]
  {
    int r    = tid >> 1;         // k row 0..127
    int half = tid & 1;          // which 8-col half
    union { uint4 q; bf16_t e[8]; } ld;
    ld.q = *(const uint4*)(B + (size_t)r * N + (blockIdx.y << 4) + half * 8);
#pragma unroll
    for (int ci = 0; ci < 8; ++ci)
      Bt[(half * 8 + ci) * BT_STRIDE + r] = ld.e[ci];
  }
  __syncthreads();

  const int lane  = tid & 31;
  const int wave  = tid >> 5;
  const int row0  = (blockIdx.x * 8 + wave) << 4;
  const int colL  = lane & 15;
  const int col   = (blockIdx.y << 4) + colL;
  const int kbase = (lane >> 4) << 3;          // 0 or 8
  int arow = row0 + colL;
  if (arow >= M) arow = M - 1;                 // clamp: OOB rows discarded at store
  const bf16_t* arowp = A + (size_t)arow * GK + kbase;
  const bf16_t* btp   = &Bt[colL * BT_STRIDE + kbase];

  union Frag { v16bf v; uint4 q[2]; };
  Frag af[4], bf[4];
#pragma unroll
  for (int s = 0; s < 4; ++s) {                // preload all A fragments (1 clause)
    af[s].q[0] = *(const uint4*)(arowp + s * 32);
    af[s].q[1] = *(const uint4*)(arowp + s * 32 + 16);
  }
#pragma unroll
  for (int s = 0; s < 4; ++s) {                // preload all B fragments (1 clause)
    bf[s].q[0] = *(const uint4*)(btp + s * 32);
    bf[s].q[1] = *(const uint4*)(btp + s * 32 + 16);
  }
  v8f acc = {};
#pragma unroll
  for (int s = 0; s < 4; ++s)                  // back-to-back WMMA chain
    acc = __builtin_amdgcn_wmma_f32_16x16x32_bf16(false, af[s].v, false, bf[s].v,
                                                  (short)0, acc, false, false);

  float bcol = bias ? bias[col] : 0.0f;
  float sg = 0.0f;
  if (mode == 2) sg = 1.0f / (1.0f + expf(-gate[0]));
  const int rbase = row0 + ((lane >> 4) << 3);
#pragma unroll
  for (int r = 0; r < 8; ++r) {
    int row = rbase + r;
    if (row < M) {
      float val = acc[r] + bcol;
      if (mode == 2) {
        val = sg * val + (1.0f - sg) * skipx[(size_t)row * N + col];
        val = fmaxf(val, 0.0f);
      } else if (mode == 1) {
        val = fmaxf(val, 0.0f);
      }
      C[(size_t)row * N + col] = val;
      if (Cbf) Cbf[(size_t)row * N + col] = (bf16_t)val;
    }
  }
}

// ---------- edge kernels: one wave32 per edge ----------
// alpha[e,h] = (q[dst,h,:] . kr[src,h,:]) * p_rel[h] / 8; atomicMax of per-dst max
__global__ void k_edge_alpha(const float* __restrict__ q, const float* __restrict__ kr,
                             const int* __restrict__ ei, const float* __restrict__ prel,
                             float* __restrict__ alpha, unsigned int* __restrict__ mkey, int E) {
  int widx = blockIdx.x * (blockDim.x >> 5) + (threadIdx.x >> 5);
  if (widx >= E) return;
  int lane = threadIdx.x & 31;
  int src = ei[widx];
  int dst = ei[E + widx];
  float4 qv = *(const float4*)(q  + (size_t)dst * 128 + lane * 4);
  float4 kv = *(const float4*)(kr + (size_t)src * 128 + lane * 4);
  float d = qv.x * kv.x + qv.y * kv.y + qv.z * kv.z + qv.w * kv.w;
  // reduce within each 16-lane half (lanes 0-15 = head 0, 16-31 = head 1)
  for (int m = 1; m < 16; m <<= 1) d += __shfl_xor(d, m, 16);
  if ((lane & 15) == 0) {
    int h = lane >> 4;
    float a = d * prel[h] * 0.125f;   // 1/sqrt(64)
    alpha[(size_t)widx * 2 + h] = a;
    atomicMax(&mkey[(size_t)dst * 2 + h], enc_key(a));
  }
}

// e = exp(alpha - max); accumulate per-dst sum
__global__ void k_edge_expsum(float* __restrict__ alpha, const int* __restrict__ ei,
                              const unsigned int* __restrict__ mkey, float* __restrict__ s, int E) {
  int idx = blockIdx.x * blockDim.x + threadIdx.x;
  if (idx >= 2 * E) return;
  int e = idx >> 1, h = idx & 1;
  int dst = ei[E + e];
  float m = dec_key(mkey[(size_t)dst * 2 + h]);
  if (!isfinite(m)) m = 0.0f;
  float ev = expf(alpha[idx] - m);
  alpha[idx] = ev;
  atomicAdd(&s[(size_t)dst * 2 + h], ev);
}

// agg[dst] += vr[src] * alpha_norm
__global__ void k_edge_scatter(const float* __restrict__ vr, const int* __restrict__ ei,
                               const float* __restrict__ alpha, const float* __restrict__ s,
                               float* __restrict__ agg, int E) {
  int widx = blockIdx.x * (blockDim.x >> 5) + (threadIdx.x >> 5);
  if (widx >= E) return;
  int lane = threadIdx.x & 31;
  int src = ei[widx];
  int dst = ei[E + widx];
  int h = lane >> 4;
  float coef = alpha[(size_t)widx * 2 + h] / (s[(size_t)dst * 2 + h] + 1e-16f);
  float4 vv = *(const float4*)(vr + (size_t)src * 128 + lane * 4);
  float* ap = agg + (size_t)dst * 128 + lane * 4;
  atomicAdd(ap + 0, vv.x * coef);
  atomicAdd(ap + 1, vv.y * coef);
  atomicAdd(ap + 2, vv.z * coef);
  atomicAdd(ap + 3, vv.w * coef);
}

// ---------- pooling ----------
// w[n] = mlp2(one_hot(x[n],6)); atomicMax per-graph key
__global__ void k_node_w(const float* __restrict__ x,
                         const float* __restrict__ W0, const float* __restrict__ b0,
                         const float* __restrict__ W1, const float* __restrict__ b1,
                         float* __restrict__ wnode, const int* __restrict__ batch,
                         unsigned int* __restrict__ wm, int N) {
  int n = blockIdx.x * blockDim.x + threadIdx.x;
  if (n >= N) return;
  int c = (int)x[n];
  c = c < 0 ? 0 : (c > 5 ? 5 : c);
  float acc = b1[0];
#pragma unroll
  for (int j = 0; j < 32; ++j) {
    float hj = fmaxf(W0[c * 32 + j] + b0[j], 0.0f);  // one-hot @ W0 picks row c
    acc += hj * W1[j];
  }
  wnode[n] = acc;
  atomicMax(&wm[batch[n]], enc_key(acc));
}

__global__ void k_node_expsum(float* __restrict__ wnode, const int* __restrict__ batch,
                              const unsigned int* __restrict__ wm, float* __restrict__ wsum, int N) {
  int n = blockIdx.x * blockDim.x + threadIdx.x;
  if (n >= N) return;
  int g = batch[n];
  float m = dec_key(wm[g]);
  if (!isfinite(m)) m = 0.0f;
  float ev = expf(wnode[n] - m);
  wnode[n] = ev;
  atomicAdd(&wsum[g], ev);
}

__global__ void k_pool_scatter(const float* __restrict__ h, const float* __restrict__ wnode,
                               const float* __restrict__ wsum, const int* __restrict__ batch,
                               float* __restrict__ g, int N) {
  int i = blockIdx.x * blockDim.x + threadIdx.x;
  if (i >= N * 128) return;
  int n = i >> 7, j = i & 127;
  int gi = batch[n];
  float coef = wnode[n] / (wsum[gi] + 1e-16f);
  atomicAdd(&g[(size_t)gi * 128 + j], h[i] * coef);
}

// ---------------------------------------------------------------------------
extern "C" void kernel_launch(void* const* d_in, const int* in_sizes, int n_in,
                              void* d_out, int out_size, void* d_ws, size_t ws_size,
                              hipStream_t stream) {
  (void)n_in; (void)out_size; (void)ws_size;
  int idx = 0;
  const float* x = (const float*)d_in[idx++];
  struct ConvP { const float *Wk,*bk,*Wq,*bq,*Wv,*bv,*Wa,*ba,*a_rel,*m_rel,*p_rel,*skip; };
  ConvP cp[3];
  for (int l = 0; l < 3; ++l) {
    cp[l].Wk = (const float*)d_in[idx++]; cp[l].bk = (const float*)d_in[idx++];
    cp[l].Wq = (const float*)d_in[idx++]; cp[l].bq = (const float*)d_in[idx++];
    cp[l].Wv = (const float*)d_in[idx++]; cp[l].bv = (const float*)d_in[idx++];
    cp[l].Wa = (const float*)d_in[idx++]; cp[l].ba = (const float*)d_in[idx++];
    cp[l].a_rel = (const float*)d_in[idx++]; cp[l].m_rel = (const float*)d_in[idx++];
    cp[l].p_rel = (const float*)d_in[idx++]; cp[l].skip  = (const float*)d_in[idx++];
  }
  const float* wmW0 = (const float*)d_in[idx++];
  const float* wmb0 = (const float*)d_in[idx++];
  const float* wmW1 = (const float*)d_in[idx++];
  const float* wmb1 = (const float*)d_in[idx++];
  const float* mW0  = (const float*)d_in[idx++];
  const float* mb0  = (const float*)d_in[idx++];
  const float* mW1  = (const float*)d_in[idx++];
  const float* mb1  = (const float*)d_in[idx++];
  const int ei0_idx = idx;
  const int* ei0   = (const int*)d_in[idx++];
  const int* ei1   = (const int*)d_in[idx++];
  const int* batch = (const int*)d_in[idx++];

  const int Nn = in_sizes[0];           // 20000 nodes
  const int E  = in_sizes[ei0_idx] / 2; // 320000 edges per type
  const int G  = NGRAPH;

  // ---- workspace bump allocator ----
  char* wsp = (char*)d_ws;
  size_t off = 0;
  auto alloc = [&](size_t bytes) -> void* {
    void* p = wsp + off;
    off += (bytes + 255) & ~(size_t)255;
    return p;
  };
  float*  h    = (float*)alloc((size_t)Nn * 128 * 4);
  bf16_t* hbf  = (bf16_t*)alloc((size_t)Nn * 128 * 2);
  float*  kf   = (float*)alloc((size_t)Nn * 128 * 4);
  float*  qf   = (float*)alloc((size_t)Nn * 128 * 4);
  float*  vf   = (float*)alloc((size_t)Nn * 128 * 4);
  bf16_t* kbf  = (bf16_t*)alloc((size_t)Nn * 128 * 2);
  bf16_t* vbf  = (bf16_t*)alloc((size_t)Nn * 128 * 2);
  float*  kr   = (float*)alloc((size_t)Nn * 128 * 4);
  float*  vr   = (float*)alloc((size_t)Nn * 128 * 4);
  float*  agg  = (float*)alloc((size_t)Nn * 128 * 4);
  bf16_t* gabf = (bf16_t*)alloc((size_t)Nn * 128 * 2);
  float*  alpha = (float*)alloc((size_t)E * 2 * 4);
  unsigned int* mkey = (unsigned int*)alloc((size_t)Nn * 2 * 4);
  float*  ssum = (float*)alloc((size_t)Nn * 2 * 4);
  bf16_t* Wkbf = (bf16_t*)alloc(128 * 128 * 2);
  bf16_t* Wqbf = (bf16_t*)alloc(128 * 128 * 2);
  bf16_t* Wvbf = (bf16_t*)alloc(128 * 128 * 2);
  bf16_t* Wabf = (bf16_t*)alloc(128 * 128 * 2);
  bf16_t* relA = (bf16_t*)alloc(128 * 128 * 2);
  bf16_t* relM = (bf16_t*)alloc(128 * 128 * 2);
  float*  wnode = (float*)alloc((size_t)Nn * 4);
  unsigned int* wmk = (unsigned int*)alloc(G * 4);
  float*  wsum = (float*)alloc(G * 4);
  float*  gp   = (float*)alloc((size_t)G * 128 * 4);
  bf16_t* gpbf = (bf16_t*)alloc((size_t)G * 128 * 2);
  float*  t1   = (float*)alloc(128 * 128 * 4);
  bf16_t* t1bf = (bf16_t*)alloc(128 * 128 * 2);
  bf16_t* mW0bf = (bf16_t*)alloc(128 * 128 * 2);
  bf16_t* mW1bf = (bf16_t*)alloc(128 * 256 * 2);

  const int NE = Nn * 128;
  auto blocks = [](int n, int bs) { return (n + bs - 1) / bs; };
  auto cvt = [&](const float* a, bf16_t* o, int n) {
    k_cvt_bf16<<<blocks(n, 256), 256, 0, stream>>>(a, o, n);
  };
  auto zero = [&](void* p, int nwords) {
    k_zero32<<<blocks(nwords, 256), 256, 0, stream>>>((unsigned int*)p, nwords);
  };
  auto gemm = [&](const bf16_t* A, const bf16_t* B, const float* bias,
                  const float* skipx, const float* gate,
                  float* C, bf16_t* Cbf, int M, int Nc, int mode) {
    dim3 g((M + 127) / 128, Nc / 16);
    k_gemm_wmma<<<g, 256, 0, stream>>>(A, B, bias, skipx, gate, C, Cbf, M, Nc, mode);
  };

  const int* eis[2] = {ei0, ei1};
  const int ewarp_blocks = blocks(E, 8);       // 8 waves/block, 1 wave/edge

  for (int l = 0; l < 3; ++l) {
    ConvP& P = cp[l];
    if (l == 0) {
      k_layer1_kqv<<<blocks(NE, 256), 256, 0, stream>>>(
          x, P.Wk, P.bk, P.Wq, P.bq, P.Wv, P.bv, kf, qf, vf, kbf, vbf, Nn);
    } else {
      cvt(P.Wk, Wkbf, 128 * 128);
      cvt(P.Wq, Wqbf, 128 * 128);
      cvt(P.Wv, Wvbf, 128 * 128);
      gemm(hbf, Wkbf, P.bk, nullptr, nullptr, kf, kbf, Nn, 128, 0);
      gemm(hbf, Wqbf, P.bq, nullptr, nullptr, qf, nullptr, Nn, 128, 0);
      gemm(hbf, Wvbf, P.bv, nullptr, nullptr, vf, vbf, Nn, 128, 0);
    }
    cvt(P.Wa, Wabf, 128 * 128);
    zero(agg, NE);
    for (int t = 0; t < 2; ++t) {
      k_blockdiag<<<64, 256, 0, stream>>>(P.a_rel + t * 2 * 64 * 64, relA);
      k_blockdiag<<<64, 256, 0, stream>>>(P.m_rel + t * 2 * 64 * 64, relM);
      gemm(kbf, relA, nullptr, nullptr, nullptr, kr, nullptr, Nn, 128, 0);
      gemm(vbf, relM, nullptr, nullptr, nullptr, vr, nullptr, Nn, 128, 0);
      zero(mkey, Nn * 2);
      zero(ssum, Nn * 2);
      k_edge_alpha<<<ewarp_blocks, 256, 0, stream>>>(qf, kr, eis[t], P.p_rel + t * 2,
                                                     alpha, mkey, E);
      k_edge_expsum<<<blocks(2 * E, 256), 256, 0, stream>>>(alpha, eis[t], mkey, ssum, E);
      k_edge_scatter<<<ewarp_blocks, 256, 0, stream>>>(vr, eis[t], alpha, ssum, agg, E);
    }
    k_gelu_bf16<<<blocks(NE, 256), 256, 0, stream>>>(agg, gabf, NE);
    // layer 1: din(1) != dout(128) -> no skip, just relu; layers 2,3: gated skip + relu
    int mode = (l == 0) ? 1 : 2;
    gemm(gabf, Wabf, P.ba, h, P.skip, h, hbf, Nn, 128, mode);
  }

  // ---- attention pooling + final MLP ----
  zero(wmk, G);
  zero(wsum, G);
  zero(gp, G * 128);
  k_node_w<<<blocks(Nn, 256), 256, 0, stream>>>(x, wmW0, wmb0, wmW1, wmb1,
                                                wnode, batch, wmk, Nn);
  k_node_expsum<<<blocks(Nn, 256), 256, 0, stream>>>(wnode, batch, wmk, wsum, Nn);
  k_pool_scatter<<<blocks(NE, 256), 256, 0, stream>>>(h, wnode, wsum, batch, gp, Nn);
  cvt(gp, gpbf, G * 128);
  cvt(mW0, mW0bf, 128 * 128);
  cvt(mW1, mW1bf, 128 * 256);
  gemm(gpbf, mW0bf, mb0, nullptr, nullptr, t1, t1bf, G, 128, 1);          // relu
  gemm(t1bf, mW1bf, mb1, nullptr, nullptr, (float*)d_out, nullptr, G, 256, 0);
}
